// MultiHeadAttention_66855460929969
// MI455X (gfx1250) — matrix operand order repaired and checked by reading, not compiled, not run on hardware
//
#include <hip/hip_runtime.h>

// Problem constants (from reference): B=2, S=2048, D=1024, H=16, DK=64
#define BB    2
#define SS    2048
#define DD    1024
#define HH    16
#define DKK   64
#define MROWS (BB * SS)   // 4096 rows for the projection GEMMs

typedef __bf16 bf16;
typedef __bf16 v16bf __attribute__((ext_vector_type(16)));
typedef float  v8f   __attribute__((ext_vector_type(8)));

union FragBF { uint4 u[2]; v16bf v; };
union Pack8  { uint4 u;    bf16 h[8]; };

// Load bf16 [p, p+8) and [p+16, p+24) (16B-aligned) into a WMMA fragment reg set.
static __device__ __forceinline__ v16bf load_frag_contig(const bf16* p) {
    FragBF f;
    f.u[0] = *(const uint4*)(p);
    f.u[1] = *(const uint4*)(p + 16);
    return f.v;
}

// Async DMA: copy one 128-byte row (64 bf16) global -> LDS as 8x b128.
// Uses CDNA5 GLOBAL_LOAD_ASYNC_TO_LDS_B128 (tracked by ASYNCcnt, in-order).
static __device__ __forceinline__ void async_copy_row128(const bf16* g, uint32_t lds_off) {
    #pragma unroll
    for (int cc = 0; cc < 8; ++cc) {
        asm volatile("global_load_async_to_lds_b128 %0, %1, off"
                     :: "v"(lds_off + cc * 16), "v"(g + cc * 8)
                     : "memory");
    }
}

// ---------------------------------------------------------------------------
// Kernel 1: P = relu(X @ W + b), output bf16.  grid.z in {0,1,2} selects Q/K/V.
// Q,K stored row-major [B*S, D].  V stored TRANSPOSED per (batch, head):
// Vt[((b*H + h)*DK + dim) * S + key], so the attention PV B-fragments become
// contiguous global loads (zero LDS transpose work in the flash inner loop).
// Block = 128 threads (4 waves), tile 64x64, K step 32; W tile staged
// transposed bf16 in LDS, shared by the 4 waves.
// ---------------------------------------------------------------------------
__global__ __launch_bounds__(128) void qkv_proj_kernel(
    const float* __restrict__ xq, const float* __restrict__ xk, const float* __restrict__ xv,
    const float* __restrict__ wq, const float* __restrict__ wk, const float* __restrict__ wv,
    const float* __restrict__ bq, const float* __restrict__ bk, const float* __restrict__ bv,
    bf16* __restrict__ oq, bf16* __restrict__ ok, bf16* __restrict__ ovt)
{
    const int z = blockIdx.z;
    const float* X  = (z == 0) ? xq : (z == 1) ? xk : xv;
    const float* W  = (z == 0) ? wq : (z == 1) ? wk : wv;
    const float* Bs = (z == 0) ? bq : (z == 1) ? bk : bv;

    const int tid  = threadIdx.x;
    const int lane = tid & 31;
    const int wave = tid >> 5;
    const int ln16 = lane & 15;
    const bool hi  = lane >= 16;

    const int colBase = blockIdx.x * 64;
    const int rowBase = blockIdx.y * 64;
    const int wrow    = rowBase + wave * 16;

    __shared__ __align__(16) bf16 WT[64 * 32];   // W^T tile: [n][k], bf16

    v8f acc[4];
    #pragma unroll
    for (int f = 0; f < 4; ++f)
        #pragma unroll
        for (int j = 0; j < 8; ++j) acc[f][j] = 0.0f;

    const int aRow  = wrow + ln16;
    const int aBase = hi ? 8 : 0;     // 16-bit A-matrix lane layout
    const int bKs   = hi ? 8 : 0;     // 16-bit B-matrix lane layout

    for (int k0 = 0; k0 < DD; k0 += 32) {
        __syncthreads();
        for (int idx = tid; idx < 32 * 64; idx += 128) {
            int r = idx >> 6;    // k within tile
            int c = idx & 63;    // n within tile
            WT[c * 32 + r] = (bf16)W[(size_t)(k0 + r) * DD + colBase + c];
        }
        __syncthreads();

        v16bf a;
        const float* xr = X + (size_t)aRow * DD + k0 + aBase;
        #pragma unroll
        for (int j = 0; j < 8; ++j) a[j] = (bf16)xr[j];
        #pragma unroll
        for (int j = 0; j < 8; ++j) a[8 + j] = (bf16)xr[16 + j];

        #pragma unroll
        for (int f = 0; f < 4; ++f) {
            v16bf bfrag = load_frag_contig(&WT[(f * 16 + ln16) * 32 + bKs]);
            acc[f] = __builtin_amdgcn_wmma_f32_16x16x32_bf16(
                false, a, false, bfrag, (short)0, acc[f], false, false);
        }
    }

    // Epilogue: bias + relu.
    if (z == 2) {
        // V: transposed store.  A lane's 8 accumulator rows are 8 consecutive
        // keys at fixed dim -> one packed 16B store per fragment.
        const int row0 = wrow + (hi ? 8 : 0);
        const int bb   = row0 >> 11;          // / SS
        const int s0   = row0 & (SS - 1);
        #pragma unroll
        for (int f = 0; f < 4; ++f) {
            const int col = colBase + f * 16 + ln16;   // global dim in [0,1024)
            const int hh  = col >> 6;                  // head
            const int dk  = col & 63;                  // dim within head
            const float bias = Bs[col];
            Pack8 pk;
            #pragma unroll
            for (int j = 0; j < 8; ++j) {
                float v = acc[f][j] + bias;
                v = v > 0.0f ? v : 0.0f;
                pk.h[j] = (bf16)v;
            }
            *(uint4*)(ovt + ((size_t)(bb * HH + hh) * DKK + dk) * SS + s0) = pk.u;
        }
    } else {
        bf16* O = (z == 0) ? oq : ok;
        #pragma unroll
        for (int f = 0; f < 4; ++f) {
            const int col = colBase + f * 16 + ln16;
            const float bias = Bs[col];
            #pragma unroll
            for (int j = 0; j < 8; ++j) {
                const int row = wrow + j + (hi ? 8 : 0);
                float v = acc[f][j] + bias;
                v = v > 0.0f ? v : 0.0f;
                O[(size_t)row * DD + col] = (bf16)v;
            }
        }
    }
}

// ---------------------------------------------------------------------------
// Kernel 2: causal flash attention.  One wave per (batch, head, 32-row q-tile).
// Per 32-key tile: 8 WMMAs for S = Q.K^T (K frags reused across q-halves) and
// 8 WMMAs for O += P.V.  K tiles are double-buffered in LDS via
// global_load_async_to_lds_b128 + s_wait_asynccnt software pipelining; the
// next V tile is touched with global_prefetch_b8.  Softmax: lane == row,
// running max/denominator in registers.  Causal mask applied analytically.
// ---------------------------------------------------------------------------
__global__ __launch_bounds__(32) void flash_attn_kernel(
    const bf16* __restrict__ Qb, const bf16* __restrict__ Kb,
    const bf16* __restrict__ Vt, float* __restrict__ Out)
{
    const int lane = threadIdx.x;
    const int ln16 = lane & 15;
    const bool hi  = lane >= 16;
    const int hi8  = hi ? 8 : 0;
    const int qt   = blockIdx.x;       // 0..63 (32-row q-tiles)
    const int h    = blockIdx.y;
    const int b    = blockIdx.z;
    const int qbase = qt * 32;

    __shared__ __align__(16) bf16  sK[2][32 * 64];   // double-buffered K tile (2x4KB)
    __shared__ __align__(16) float sS[32 * 32];
    __shared__ __align__(16) bf16  sP[32 * 32];
    __shared__ float scorr[32], sl[32];

    const int aBase = hi ? 8 : 0;
    const int bKs   = hi ? 8 : 0;

    // Q fragments: 2 q-halves x 2 dim-halves (fixed over the whole k loop).
    v16bf aq[2][2];
    #pragma unroll
    for (int qh = 0; qh < 2; ++qh) {
        const bf16* qrow =
            Qb + (size_t)(b * SS + qbase + qh * 16 + ln16) * DD + h * DKK;
        #pragma unroll
        for (int kh = 0; kh < 2; ++kh) {
            FragBF f;
            f.u[0] = *(const uint4*)(qrow + kh * 32 + aBase);
            f.u[1] = *(const uint4*)(qrow + kh * 32 + 16 + aBase);
            aq[qh][kh] = f.v;
        }
    }

    v8f o[2][4];
    #pragma unroll
    for (int qh = 0; qh < 2; ++qh)
        #pragma unroll
        for (int f = 0; f < 4; ++f)
            #pragma unroll
            for (int j = 0; j < 8; ++j) o[qh][f][j] = 0.0f;

    float m_i = -3.0e38f;   // running max for row (qbase + lane)
    float l_i = 0.0f;       // running denominator

    const bf16* vt_head = Vt + (size_t)(b * HH + h) * DKK * SS;
    const uint32_t sk_base = (uint32_t)(uintptr_t)&sK[0][0];
    const float scale = 0.125f;          // 1/sqrt(DK)
    const int nkt = qt + 1;              // causal tile count

    // Preload K tile 0 into buffer 0 (one 128B row per lane).
    async_copy_row128(Kb + (size_t)(b * SS + lane) * DD + h * DKK,
                      sk_base + lane * 128);

    for (int kt = 0; kt < nkt; ++kt) {
        const int kbase = kt * 32;
        const int cur = kt & 1;

        if (kt + 1 < nkt) {
            // Software pipeline: launch DMA for tile kt+1, prefetch next V,
            // then wait for the oldest 8 async ops (tile kt) to land.
            async_copy_row128(
                Kb + (size_t)(b * SS + kbase + 32 + lane) * DD + h * DKK,
                sk_base + (cur ^ 1) * (32 * 64 * 2) + lane * 128);
            const bf16* vpre = vt_head + (size_t)(lane * 2) * SS + kbase + 32;
            __builtin_prefetch(vpre, 0, 3);
            __builtin_prefetch(vpre + SS, 0, 3);
            asm volatile("s_wait_asynccnt 0x8" ::: "memory");
        } else {
            asm volatile("s_wait_asynccnt 0x0" ::: "memory");
        }

        // S(32x32) = Q(32x64) . K^T : 8 WMMAs, K frags shared across q-halves.
        v8f cs[2][2];
        #pragma unroll
        for (int qh = 0; qh < 2; ++qh)
            #pragma unroll
            for (int nh = 0; nh < 2; ++nh)
                #pragma unroll
                for (int j = 0; j < 8; ++j) cs[qh][nh][j] = 0.0f;

        const bf16* skt = &sK[cur][0];
        #pragma unroll
        for (int nh = 0; nh < 2; ++nh) {
            const bf16* krow = skt + (nh * 16 + ln16) * 64;
            #pragma unroll
            for (int kh = 0; kh < 2; ++kh) {
                v16bf bfragK = load_frag_contig(krow + kh * 32 + bKs);
                #pragma unroll
                for (int qh = 0; qh < 2; ++qh)
                    cs[qh][nh] = __builtin_amdgcn_wmma_f32_16x16x32_bf16(
                        false, aq[qh][kh], false, bfragK, (short)0,
                        cs[qh][nh], false, false);
            }
        }

        __syncthreads();
        // Scaled + causally-masked scores -> LDS.
        #pragma unroll
        for (int nh = 0; nh < 2; ++nh) {
            const int col = nh * 16 + ln16;
            const int kg  = kbase + col;
            #pragma unroll
            for (int qh = 0; qh < 2; ++qh) {
                #pragma unroll
                for (int j = 0; j < 8; ++j) {
                    const int row = qh * 16 + j + hi8;
                    float v = cs[qh][nh][j] * scale;
                    if (kg > qbase + row) v = -1.0e10f;   // NEG from reference
                    sS[row * 32 + col] = v;
                }
            }
        }
        __syncthreads();

        // Online softmax: lane owns row (qbase + lane), 32 columns.
        {
            const float4* srow = (const float4*)&sS[lane * 32];
            float4 rv[8];
            #pragma unroll
            for (int r = 0; r < 8; ++r) rv[r] = srow[r];
            float tmax = -3.0e38f;
            #pragma unroll
            for (int r = 0; r < 8; ++r)
                tmax = fmaxf(tmax,
                             fmaxf(fmaxf(rv[r].x, rv[r].y), fmaxf(rv[r].z, rv[r].w)));
            const float mnew = fmaxf(m_i, tmax);
            const float corr = __expf(m_i - mnew);
            float lsum = 0.0f;
            uint4* prow = (uint4*)&sP[lane * 32];
            #pragma unroll
            for (int r = 0; r < 8; r += 2) {
                Pack8 pk;
                #pragma unroll
                for (int t = 0; t < 2; ++t) {
                    const float4 v4 = rv[r + t];
                    float e0 = __expf(v4.x - mnew), e1 = __expf(v4.y - mnew);
                    float e2 = __expf(v4.z - mnew), e3 = __expf(v4.w - mnew);
                    lsum += e0 + e1 + e2 + e3;
                    pk.h[t * 4 + 0] = (bf16)e0; pk.h[t * 4 + 1] = (bf16)e1;
                    pk.h[t * 4 + 2] = (bf16)e2; pk.h[t * 4 + 3] = (bf16)e3;
                }
                prow[r >> 1] = pk.u;
            }
            l_i = l_i * corr + lsum;
            m_i = mnew;
            scorr[lane] = corr;
        }
        __syncthreads();

        // Rescale accumulators by per-row correction.
        #pragma unroll
        for (int qh = 0; qh < 2; ++qh)
            #pragma unroll
            for (int j = 0; j < 8; ++j) {
                const float cf = scorr[qh * 16 + j + hi8];
                #pragma unroll
                for (int f = 0; f < 4; ++f) o[qh][f][j] *= cf;
            }

        // P fragments.
        v16bf ap[2];
        #pragma unroll
        for (int qh = 0; qh < 2; ++qh) {
            FragBF f;
            const bf16* prow = sP + (size_t)(qh * 16 + ln16) * 32;
            f.u[0] = *(const uint4*)(prow + aBase);
            f.u[1] = *(const uint4*)(prow + 16 + aBase);
            ap[qh] = f.v;
        }

        // O(32x64) += P(32x32) . V(32x64): V^T frags contiguous from global.
        #pragma unroll
        for (int f4 = 0; f4 < 4; ++f4) {
            v16bf bfragV = load_frag_contig(
                vt_head + (size_t)(f4 * 16 + ln16) * SS + kbase + bKs);
            #pragma unroll
            for (int qh = 0; qh < 2; ++qh)
                o[qh][f4] = __builtin_amdgcn_wmma_f32_16x16x32_bf16(
                    false, ap[qh], false, bfragV, (short)0, o[qh][f4], false, false);
        }
        __syncthreads();
    }

    sl[lane] = l_i;
    __syncthreads();

    // Final: divide by denominator, store fp32 (heads = contiguous 64-col slices).
    #pragma unroll
    for (int qh = 0; qh < 2; ++qh)
        #pragma unroll
        for (int j = 0; j < 8; ++j) {
            const int row = qh * 16 + j + hi8;
            const float inv = 1.0f / sl[row];
            #pragma unroll
            for (int f4 = 0; f4 < 4; ++f4) {
                const int col = h * DKK + f4 * 16 + ln16;
                Out[(size_t)(b * SS + qbase + row) * DD + col] = o[qh][f4][j] * inv;
            }
        }
}

// ---------------------------------------------------------------------------
extern "C" void kernel_launch(void* const* d_in, const int* in_sizes, int n_in,
                              void* d_out, int out_size, void* d_ws, size_t ws_size,
                              hipStream_t stream) {
    (void)in_sizes; (void)n_in; (void)out_size; (void)ws_size;
    const float* q  = (const float*)d_in[0];
    const float* k  = (const float*)d_in[1];
    const float* v  = (const float*)d_in[2];
    // d_in[3] = causal mask: applied analytically, never read (saves 33.5 MB).
    const float* Wq = (const float*)d_in[4];
    const float* bq = (const float*)d_in[5];
    const float* Wk = (const float*)d_in[6];
    const float* bk = (const float*)d_in[7];
    const float* Wv = (const float*)d_in[8];
    const float* bv = (const float*)d_in[9];

    bf16* Qb = (bf16*)d_ws;                    // 4096x1024 bf16 = 8 MB
    bf16* Kb = Qb + (size_t)MROWS * DD;        // +8 MB
    bf16* Vt = Kb + (size_t)MROWS * DD;        // +8 MB (V stored transposed)

    dim3 pgrid(DD / 64, MROWS / 64, 3);
    qkv_proj_kernel<<<pgrid, 128, 0, stream>>>(q, k, v, Wq, Wk, Wv, bq, bk, bv,
                                               Qb, Kb, Vt);

    dim3 agrid(SS / 32, HH, BB);
    flash_attn_kernel<<<agrid, 32, 0, stream>>>(Qb, Kb, Vt, (float*)d_out);
}
